// SoftNCutsLoss_38611755991341
// MI455X (gfx1250) — matrix operand order repaired and checked by reading, not compiled, not run on hardware
//
#include <hip/hip_runtime.h>
#include <hip/hip_bf16.h>

typedef __attribute__((ext_vector_type(16))) _Float16 v16h;
typedef __attribute__((ext_vector_type(8)))  _Float16 v8h;
typedef __attribute__((ext_vector_type(8)))  float    v8f;

#define NB 2
#define NK 5
#define DD 96
#define HH 96
#define WW 96
#define NVOX ((size_t)DD * HH * WW)          /* 884736 */
#define HP 104                                /* 4 + 96 + 4   */
#define WPAD 112                              /* 4 + 96 + 12  */
#define PLANE ((size_t)HP * WPAD)             /* 11648        */
#define FVOL  ((size_t)DD * PLANE)            /* 1118208      */
#define BTAB_HALVES (81 * 32 * 16)            /* 41472        */
#define BTAB_BYTES  (BTAB_HALVES * 2)         /* 82944        */
#define SUMS_OFF    BTAB_BYTES
#define FIELDS_OFF  (BTAB_BYTES + 512)        /* 83456, 256-aligned */
/* sums layout (floats): [0..9]=sumL  [10..19]=sumXL  [20..24]=numer  [25..29]=denom */

// ---------------------------------------------------------------------------
// 1) Init: zero accumulators; build Toeplitz B-fragments for all 81 (kd,kh)
//    in the per-lane VGPR layout of the 32x16 f16 B operand.
// ---------------------------------------------------------------------------
__global__ __launch_bounds__(256) void sn_init(float* sums, _Float16* btab) {
  int tid = blockIdx.x * blockDim.x + threadIdx.x;
  if (tid < 30) sums[tid] = 0.0f;
  if (tid < BTAB_HALVES) {
    int i  = tid & 15;          // half index within lane (K mod 16)
    int l  = (tid >> 4) & 31;   // lane
    int p  = tid >> 9;          // (kd*9+kh)
    int kd = p / 9 - 4;
    int kh = p % 9 - 4;
    int n  = l & 15;                     // output column
    int kk = (l < 16) ? i : (16 + i);    // K row
    int j  = kk - n;                     // tap index along W
    float v = 0.0f;
    if (j >= 0 && j <= 8) {
      int kw = j - 4;
      float s = (float)(kd * kd + kh * kh + kw * kw) * 0.2f;  // /SIGMA_X
      v = __expf(-0.5f * s * s);
    }
    btab[tid] = (_Float16)v;
  }
}

// ---------------------------------------------------------------------------
// 2) Per-(b,k) reduction: sum(labels), sum(inputs*labels)
// ---------------------------------------------------------------------------
#define RED_SLICES 64
__global__ __launch_bounds__(256) void sn_reduce(const float* __restrict__ inputs,
                                                 const float* __restrict__ labels,
                                                 float* sums) {
  int bk = blockIdx.x / RED_SLICES;
  int sl = blockIdx.x % RED_SLICES;
  int b  = bk / NK;
  const float* lab = labels + (size_t)bk * NVOX;
  const float* inp = inputs + (size_t)b  * NVOX;
  float sL = 0.0f, sXL = 0.0f;
  for (size_t i = (size_t)sl * 256 + threadIdx.x; i < NVOX; i += (size_t)RED_SLICES * 256) {
    float l = lab[i];
    sL += l;
    sXL += l * inp[i];
  }
  __shared__ float sh[512];
  sh[threadIdx.x] = sL;
  sh[256 + threadIdx.x] = sXL;
  __syncthreads();
  for (int s = 128; s > 0; s >>= 1) {
    if ((int)threadIdx.x < s) {
      sh[threadIdx.x] += sh[threadIdx.x + s];
      sh[256 + threadIdx.x] += sh[256 + threadIdx.x + s];
    }
    __syncthreads();
  }
  if (threadIdx.x == 0) {
    atomicAdd(&sums[bk], sh[0]);
    atomicAdd(&sums[10 + bk], sh[256]);
  }
}

// ---------------------------------------------------------------------------
// 3) Field build: w = exp(-((x-mean)^2)^2), lw = label*w; f16, zero-padded
//    fields[bk][f][d][hp][wp], f=0 -> lw, f=1 -> w
// ---------------------------------------------------------------------------
__global__ __launch_bounds__(256) void sn_fields(const float* __restrict__ inputs,
                                                 const float* __restrict__ labels,
                                                 const float* __restrict__ sums,
                                                 _Float16* __restrict__ fields) {
  const size_t total = (size_t)NB * NK * DD * HP * WPAD;
  for (size_t idx = (size_t)blockIdx.x * blockDim.x + threadIdx.x; idx < total;
       idx += (size_t)gridDim.x * blockDim.x) {
    size_t t = idx;
    int wp = (int)(t % WPAD); t /= WPAD;
    int hp = (int)(t % HP);   t /= HP;
    int d  = (int)(t % DD);   t /= DD;
    int bk = (int)t;
    int b  = bk / NK;
    _Float16 wv = (_Float16)0.0f, lwv = (_Float16)0.0f;
    int h = hp - 4, w = wp - 4;
    if (h >= 0 && h < HH && w >= 0 && w < WW) {
      size_t vox = ((size_t)d * HH + h) * WW + w;
      float x   = inputs[(size_t)b * NVOX + vox];
      float lab = labels[(size_t)bk * NVOX + vox];
      const float invN = 1.0f / (float)NVOX;
      float mean = (sums[10 + bk] * invN) / (sums[bk] * invN + 1e-5f);
      float dsq = (x - mean) * (x - mean);
      float wgt = __expf(-dsq * dsq);              // SIGMA_I = 1
      wv  = (_Float16)wgt;
      lwv = (_Float16)(lab * wgt);
    }
    size_t base = (size_t)bk * 2 * FVOL + (size_t)d * PLANE + (size_t)hp * WPAD + wp;
    fields[base]        = lwv;   // f = 0
    fields[base + FVOL] = wv;    // f = 1
  }
}

// ---------------------------------------------------------------------------
// 4) Conv + dot.  One wave per 16x16 output tile at fixed depth.
//    Per kd-plane the wave stages a 2-field 24x32 f16 A-tile into LDS with
//    global_load_async_to_lds_b128 (double-buffered, ASYNCcnt), then runs the
//    9 (kh) Toeplitz GEMMs with A from ds_load_b128 and B from the L2-hot
//    table:  D(16x16) += A(16x32) x B(32x16) per (kd,kh), per field.
// ---------------------------------------------------------------------------
#define TROWS 24
#define TCOLS 32
#define TSLICE (2 * TROWS * TCOLS)   /* halves per (wave,buffer): 1536 */

__device__ __forceinline__ void sn_issue_plane(const _Float16* __restrict__ fbase,
                                               size_t planeoff, int h0, int w0,
                                               _Float16* ldsbuf, int lane) {
#pragma unroll
  for (int it = 0; it < 6; ++it) {
    int c     = it * 32 + lane;       // 192 16-byte chunks: 2 fields x 24 rows x 4
    int field = c / 96;
    int rr    = (c % 96) >> 2;
    int ch    = c & 3;
    const _Float16* g = fbase + (size_t)field * FVOL + planeoff +
                        (size_t)(h0 + rr) * WPAD + (size_t)(w0 + ch * 8);
    _Float16* l = ldsbuf + ((field * TROWS + rr) * TCOLS + ch * 8);
    unsigned loff = (unsigned)(size_t)l;                 // LDS byte offset (low dword)
    unsigned long long gaddr = (unsigned long long)(size_t)g;
    asm volatile("global_load_async_to_lds_b128 %0, %1, off"
                 :: "v"(loff), "v"(gaddr) : "memory");
  }
}

__global__ __launch_bounds__(256) void sn_conv(const float* __restrict__ labels,
                                               const _Float16* __restrict__ fields,
                                               const _Float16* __restrict__ btab,
                                               float* sums) {
  __shared__ _Float16 tile[8][2][TSLICE];   // 49152 bytes
  int wslot = threadIdx.x >> 5;
  int lane  = threadIdx.x & 31;

  int t = blockIdx.x * (blockDim.x >> 5) + wslot;
  int tw = t % 6;  t /= 6;
  int th = t % 6;  t /= 6;
  int d  = t % DD; t /= DD;
  int k  = t % NK; t /= NK;
  int b  = t;
  if (b >= NB) return;  // wave-uniform guard

  const int h0 = th * 16, w0 = tw * 16;
  const int m     = lane & 15;               // A row index within tile
  const int kbase = (lane < 16) ? 0 : 16;    // A operand K striping
  const _Float16* fbase = fields + (size_t)(b * NK + k) * 2 * FVOL;

  v8f acc0 = {};  // conv(labels*w)
  v8f acc1 = {};  // conv(w)

  int kdlo = (d >= 4) ? 0 : (4 - d);
  int kdhi = (99 - d < 8) ? (99 - d) : 8;

  _Float16* buf0 = &tile[wslot][0][0];
  _Float16* buf1 = &tile[wslot][1][0];

  // preload first plane
  sn_issue_plane(fbase, (size_t)(d + kdlo - 4) * PLANE, h0, w0, buf0, lane);

  for (int kd = kdlo; kd <= kdhi; ++kd) {
    _Float16* cur = ((kd - kdlo) & 1) ? buf1 : buf0;
    _Float16* nxt = ((kd - kdlo) & 1) ? buf0 : buf1;
    if (kd < kdhi) {
      sn_issue_plane(fbase, (size_t)(d + kd - 3) * PLANE, h0, w0, nxt, lane);
      asm volatile("s_wait_asynccnt 0x6" ::: "memory");  // current plane complete
    } else {
      asm volatile("s_wait_asynccnt 0x0" ::: "memory");
    }
#pragma unroll
    for (int kh = 0; kh < 9; ++kh) {
      // B fragment: per-lane Toeplitz, 16 halves = 2 x b128 (L2-resident table)
      const _Float16* bt = btab + (((size_t)(kd * 9 + kh) * 32) + lane) * 16;
      v8h blo = *(const v8h*)bt;
      v8h bhi = *(const v8h*)(bt + 8);
      v16h bm = __builtin_shufflevector(blo, bhi, 0, 1, 2, 3, 4, 5, 6, 7,
                                        8, 9, 10, 11, 12, 13, 14, 15);
      // A fragments from LDS: row m+kh, halves [kbase..+15]
      const _Float16* r0 = cur + ((m + kh) * TCOLS + kbase);              // lw
      const _Float16* r1 = cur + ((TROWS + m + kh) * TCOLS + kbase);      // w
      v8h a0lo = *(const v8h*)r0;
      v8h a0hi = *(const v8h*)(r0 + 8);
      v8h a1lo = *(const v8h*)r1;
      v8h a1hi = *(const v8h*)(r1 + 8);
      v16h am0 = __builtin_shufflevector(a0lo, a0hi, 0, 1, 2, 3, 4, 5, 6, 7,
                                         8, 9, 10, 11, 12, 13, 14, 15);
      v16h am1 = __builtin_shufflevector(a1lo, a1hi, 0, 1, 2, 3, 4, 5, 6, 7,
                                         8, 9, 10, 11, 12, 13, 14, 15);
      acc0 = __builtin_amdgcn_wmma_f32_16x16x32_f16(false, am0, false, bm,
                                                    (short)0, acc0, false, false);
      acc1 = __builtin_amdgcn_wmma_f32_16x16x32_f16(false, am1, false, bm,
                                                    (short)0, acc1, false, false);
    }
  }

  // dot with labels: C/D layout -> VGPR g holds (M = g or g+8, N = lane&15)
  const int M0 = (lane < 16) ? 0 : 8;
  const int n  = lane & 15;
  const float* lp = labels + (((size_t)(b * NK + k) * DD + d) * HH) * WW;
  float pn = 0.0f, pd = 0.0f;
#pragma unroll
  for (int g = 0; g < 8; ++g) {
    float lab = lp[(size_t)(h0 + g + M0) * WW + (w0 + n)];
    pn += lab * acc0[g];
    pd += lab * acc1[g];
  }
  for (int off = 16; off > 0; off >>= 1) {
    pn += __shfl_down(pn, off, 32);
    pd += __shfl_down(pd, off, 32);
  }
  if (lane == 0) {
    atomicAdd(&sums[20 + k], pn);
    atomicAdd(&sums[25 + k], pd);
  }
}

// ---------------------------------------------------------------------------
// 5) Finalize: out = K - sum_k |numer_k / (denom_k + 1e-6)|
// ---------------------------------------------------------------------------
__global__ void sn_final(const float* sums, float* out) {
  float loss = 0.0f;
  for (int k = 0; k < NK; ++k)
    loss += fabsf(sums[20 + k] / (sums[25 + k] + 1e-6f));
  out[0] = (float)NK - loss;
}

// ---------------------------------------------------------------------------
extern "C" void kernel_launch(void* const* d_in, const int* in_sizes, int n_in,
                              void* d_out, int out_size, void* d_ws, size_t ws_size,
                              hipStream_t stream) {
  (void)in_sizes; (void)n_in; (void)out_size; (void)ws_size;
  const float* inputs = (const float*)d_in[0];   // [2,1,96,96,96] f32
  const float* labels = (const float*)d_in[1];   // [2,5,96,96,96] f32
  float* out = (float*)d_out;

  char* ws = (char*)d_ws;
  _Float16* btab   = (_Float16*)ws;
  float*    sums   = (float*)(ws + SUMS_OFF);
  _Float16* fields = (_Float16*)(ws + FIELDS_OFF);

  sn_init<<<(BTAB_HALVES + 255) / 256, 256, 0, stream>>>(sums, btab);
  sn_reduce<<<NB * NK * RED_SLICES, 256, 0, stream>>>(inputs, labels, sums);
  sn_fields<<<8192, 256, 0, stream>>>(inputs, labels, sums, fields);
  // tiles: 2 * 5 * 96 * 6 * 6 = 34560 waves; 8 waves per 256-thread block
  sn_conv<<<34560 / 8, 256, 0, stream>>>(labels, fields, btab, sums);
  sn_final<<<1, 1, 0, stream>>>(sums, out);
}
